// Attention_27152783245463
// MI455X (gfx1250) — matrix operand order repaired and checked by reading, not compiled, not run on hardware
//
#include <hip/hip_runtime.h>

// Flash attention for [bs=2, 3*8*64, 4096] f32 qkv, 8 heads, ch=64.
// bf16 WMMA (v_wmma_f32_16x16x32_bf16) for QK^T and P*V, f32 online softmax.

typedef __attribute__((ext_vector_type(16))) __bf16 v16bf;
typedef __attribute__((ext_vector_type(8)))  float  v8f;

#define HEADS   8
#define CH      64
#define LEN     4096
#define BT      64            // query rows per workgroup
#define BS      64            // key/value cols per S-iteration
#define WAVES   4             // 4 waves x 16 query rows
#define THREADS (WAVES * 32)

// f32 -> bf16, round-to-nearest-even
__device__ __forceinline__ __bf16 f2bf(float f) {
  unsigned u = __builtin_bit_cast(unsigned, f);
  u += 0x7FFFu + ((u >> 16) & 1u);
  unsigned short h = (unsigned short)(u >> 16);
  return __builtin_bit_cast(__bf16, h);
}

// A-matrix fragment (16x32 bf16, M x K) from LDS buffer laid out [m][k], row stride 64.
// ISA 7.12.2 16-bit A layout: lane half h=lane>>4, m=lane&15;
//   VGPR 0..3 hold K = 8h + {0,1},{2,3},{4,5},{6,7};  VGPR 4..7 hold K = 16 + 8h + ...
__device__ __forceinline__ v16bf load_a_frag(const __bf16* base, int kbase, int lane) {
  union { v16bf v; unsigned u[8]; } r;
  const int m = lane & 15;
  const int h = lane >> 4;
  const unsigned* row = (const unsigned*)(base + m * 64);
  #pragma unroll
  for (int i = 0; i < 4; ++i) {
    const int k = kbase + 8 * h + 2 * i;   // always even -> 32-bit LDS reads
    r.u[i]     = row[k >> 1];
    r.u[i + 4] = row[(k + 16) >> 1];
  }
  return r.v;
}

// B-matrix fragment (32x16 bf16, K x N) from LDS buffer laid out [n][k], row stride 64.
// Lane half h=lane>>4, N=lane&15; VGPR 0..3 hold K = kbase + 8h + 0..7; VGPR 4..7 hold K = kbase + 16 + 8h + 0..7.
__device__ __forceinline__ v16bf load_b_frag(const __bf16* buf, int nbase, int kbase, int lane) {
  union { v16bf v; unsigned u[8]; } r;
  const int n  = nbase + (lane & 15);
  const int k0 = (kbase + 8 * (lane >> 4)) >> 1;
  const unsigned* row = (const unsigned*)(buf + n * 64);
  #pragma unroll
  for (int i = 0; i < 4; ++i) { r.u[i] = row[k0 + i]; r.u[i + 4] = row[k0 + 8 + i]; }
  return r.v;
}

__global__ __launch_bounds__(THREADS)
void Attention_27152783245463_kernel(const float* __restrict__ qkv,
                                     float* __restrict__ out) {
  __shared__ __bf16 Qb[BT * CH];            // [t][c]
  __shared__ __bf16 Kb[BS * CH];            // [s][c]  (B layout for QK^T)
  __shared__ __bf16 Vb[CH * BS];            // [c][s]  (B layout for P*V)
  __shared__ float  Sscr[WAVES * 16 * BS];  // per-wave f32 score tile
  __shared__ __bf16 Pb[WAVES * 16 * BS];    // per-wave bf16 probabilities
  __shared__ float  alphaS[WAVES * 16];
  __shared__ float  invL[WAVES * 16];

  const int tid  = threadIdx.x;
  const int lane = tid & 31;
  const int wave = tid >> 5;
  const int bh   = blockIdx.y;              // batch*8 + head
  const int b    = bh >> 3;
  const int h    = bh & 7;
  const int t0   = blockIdx.x * BT;

  const float scale = 0.35355339059327373f; // 64^-0.25 (applied to both Q and K)

  const float* qg = qkv + (size_t)(b * 1536 + h * 64) * LEN;
  const float* kg = qg + (size_t)512  * LEN;
  const float* vg = qg + (size_t)1024 * LEN;

  // ---- Stage Q tile: Qb[t][c] = q[c][t0+t] * scale (coalesced over t) ----
  {
    const int tl = tid & 63;
    for (int c = tid >> 6; c < CH; c += THREADS / 64)
      Qb[tl * CH + c] = f2bf(qg[(size_t)c * LEN + t0 + tl] * scale);
  }
  __syncthreads();

  // Persistent Q A-fragments for this wave's 16 query rows.
  const v16bf qa0 = load_a_frag(Qb + wave * 16 * CH, 0, lane);
  const v16bf qa1 = load_a_frag(Qb + wave * 16 * CH, 32, lane);

  v8f O[4] = {};               // 16 x 64 f32 output accumulator (4 C-frags)
  float m_run = -3.0e38f;      // running row max (lanes 0-15, row = lane)
  float l_run = 0.0f;          // running row sum

  for (int sb = 0; sb < LEN / BS; ++sb) {
    const int s0 = sb * BS;
    __syncthreads();           // protect Kb/Vb reuse across waves
    {
      const int sl = tid & 63;
      for (int c = tid >> 6; c < CH; c += THREADS / 64) {
        Kb[sl * CH + c] = f2bf(kg[(size_t)c * LEN + s0 + sl] * scale);
        Vb[c * BS + sl] = f2bf(vg[(size_t)c * LEN + s0 + sl]);
      }
      if (s0 + BS < LEN) {     // hint next K/V tile into cache (global_prefetch_b8)
        __builtin_prefetch(&kg[(size_t)(tid >> 6) * LEN + s0 + BS + sl], 0, 1);
        __builtin_prefetch(&vg[(size_t)(tid >> 6) * LEN + s0 + BS + sl], 0, 1);
      }
    }
    __syncthreads();

    // ---- S = (Q*scale)^T (K*scale): 16 x 64 tile per wave ----
    const int hh = lane >> 4;
    const int nn = lane & 15;
    #pragma unroll
    for (int n = 0; n < 4; ++n) {
      v8f c = {};
      v16bf bk0 = load_b_frag(Kb, n * 16, 0, lane);
      c = __builtin_amdgcn_wmma_f32_16x16x32_bf16(false, qa0, false, bk0, (short)0, c, false, false);
      v16bf bk1 = load_b_frag(Kb, n * 16, 32, lane);
      c = __builtin_amdgcn_wmma_f32_16x16x32_bf16(false, qa1, false, bk1, (short)0, c, false, false);
      #pragma unroll
      for (int r = 0; r < 8; ++r)
        Sscr[(wave * 16 + r + 8 * hh) * BS + n * 16 + nn] = c[r];
    }

    // ---- Online softmax over this wave's rows (lanes 0-15 own one row each) ----
    if (lane < 16) {
      const int row = wave * 16 + lane;
      float* srow = &Sscr[row * BS];
      float mx = m_run;
      #pragma unroll 8
      for (int s = 0; s < BS; ++s) mx = fmaxf(mx, srow[s]);
      const float alpha = __expf(m_run - mx);
      float sum = 0.0f;
      __bf16* prow = &Pb[row * BS];
      #pragma unroll 8
      for (int s = 0; s < BS; ++s) {
        const float p = __expf(srow[s] - mx);
        sum += p;
        prow[s] = f2bf(p);
      }
      l_run = l_run * alpha + sum;
      m_run = mx;
      alphaS[row] = alpha;
    }
    // Per-wave LDS ops are in-order; Pb/alphaS writes above are visible to this wave here.

    // ---- O = O*alpha + P * V^T ----
    const v16bf pa0 = load_a_frag(Pb + wave * 16 * BS, 0, lane);
    const v16bf pa1 = load_a_frag(Pb + wave * 16 * BS, 32, lane);
    float arow[8];
    #pragma unroll
    for (int r = 0; r < 8; ++r) arow[r] = alphaS[wave * 16 + r + 8 * hh];
    #pragma unroll
    for (int n = 0; n < 4; ++n) {
      v8f o = O[n];
      #pragma unroll
      for (int r = 0; r < 8; ++r) o[r] *= arow[r];
      v16bf bv0 = load_b_frag(Vb, n * 16, 0, lane);
      o = __builtin_amdgcn_wmma_f32_16x16x32_bf16(false, pa0, false, bv0, (short)0, o, false, false);
      v16bf bv1 = load_b_frag(Vb, n * 16, 32, lane);
      o = __builtin_amdgcn_wmma_f32_16x16x32_bf16(false, pa1, false, bv1, (short)0, o, false, false);
      O[n] = o;
    }
  }

  // ---- Epilogue: divide by row sums, store out[b, h*64+c, t] ----
  if (lane < 16) invL[wave * 16 + lane] = 1.0f / l_run;
  const int hh = lane >> 4;
  const int nn = lane & 15;
  float il[8];
  #pragma unroll
  for (int r = 0; r < 8; ++r) il[r] = invL[wave * 16 + r + 8 * hh];
  #pragma unroll
  for (int n = 0; n < 4; ++n) {
    #pragma unroll
    for (int r = 0; r < 8; ++r) {
      const int t = t0 + wave * 16 + r + 8 * hh;
      const int c = n * 16 + nn;
      out[(size_t)(bh * 64 + c) * LEN + t] = O[n][r] * il[r];
    }
  }
}

extern "C" void kernel_launch(void* const* d_in, const int* in_sizes, int n_in,
                              void* d_out, int out_size, void* d_ws, size_t ws_size,
                              hipStream_t stream) {
  (void)n_in; (void)out_size; (void)d_ws; (void)ws_size;
  const float* qkv = (const float*)d_in[0];
  float* out = (float*)d_out;
  const int bs = in_sizes[0] / (3 * HEADS * CH * LEN); // = 2
  dim3 grid(LEN / BT, bs * HEADS);
  Attention_27152783245463_kernel<<<grid, THREADS, 0, stream>>>(qkv, out);
}
